// Self_Attention_50714973831190
// MI455X (gfx1250) — matrix operand order repaired and checked by reading, not compiled, not run on hardware
//
#include <hip/hip_runtime.h>

typedef __attribute__((ext_vector_type(16))) _Float16 v16h;
typedef __attribute__((ext_vector_type(8)))  _Float16 v8h;
typedef __attribute__((ext_vector_type(8)))  float    v8f;

#define BATCH 4
#define SEQ   2048
#define DIM   1024
#define EOUT  1024

#if defined(__has_builtin)
#if __has_builtin(__builtin_amdgcn_global_load_async_to_lds_b128) && \
    __has_builtin(__builtin_amdgcn_s_wait_asynccnt)
#define HAVE_ASYNC_LDS 1
#endif
#endif

#if defined(HAVE_ASYNC_LDS)
typedef int async_v4i __attribute__((vector_size(4 * sizeof(int))));
typedef async_v4i __attribute__((address_space(1)))* async_gptr;
typedef async_v4i __attribute__((address_space(3)))* async_lptr;
#endif

static __device__ __forceinline__ v16h cat8(v8h lo, v8h hi) {
  return __builtin_shufflevector(lo, hi, 0,1,2,3,4,5,6,7,8,9,10,11,12,13,14,15);
}

// =====================================================================
// Kernel 1: QKV projection GEMM.  C = X(8192x1024) * W_z(1024x1024)
// z = blockIdx.z selects Q/K/V weight slice.  Q,K stored row-major f16;
// V stored transposed (WVt[b][e][s]) so P@V has contiguous contraction.
// Double-buffered LDS, register-staged global loads, 1 barrier / K-step.
// =====================================================================
__global__ __launch_bounds__(256) void qkv_gemm(
    const float* __restrict__ X, const float* __restrict__ Wall,
    _Float16* __restrict__ WQ, _Float16* __restrict__ WK,
    _Float16* __restrict__ WVt)
{
  __shared__ __align__(16) _Float16 As[2][128 * 40];  // A[m][k], stride 40
  __shared__ __align__(16) _Float16 Bs[2][128 * 40];  // B^T: Bs[n][k], stride 40

  const int z    = blockIdx.z;
  const float* W = Wall + (size_t)z * DIM * EOUT;
  const int m0   = blockIdx.x * 128;
  const int n0   = blockIdx.y * 128;
  const int tid  = threadIdx.x;
  const int wave = tid >> 5, lane = tid & 31;
  const int wm   = wave >> 2, wn = wave & 3;          // 2x4 wave grid
  const int lr   = lane & 15;
  const int lhi  = (lane >= 16) ? 1 : 0;

  v8f acc[4][2] = {};
  float4 ra[4], rb[4];

  const int ak  = lhi ? 8 : 0;    // A-frag K-octet base
  const int bko = lhi ? 8 : 0;    // B-frag K base

  auto loadG = [&](int kk) {
    #pragma unroll
    for (int i = 0; i < 4; ++i) {
      const int fid = tid + i * 256;
      ra[i] = *(const float4*)(X + (size_t)(m0 + (fid >> 3)) * DIM + kk + (fid & 7) * 4);
      rb[i] = *(const float4*)(W + (size_t)(kk + (fid >> 5)) * EOUT + n0 + (fid & 31) * 4);
    }
  };
  auto stage = [&](int buf) {
    #pragma unroll
    for (int i = 0; i < 4; ++i) {
      const int fid = tid + i * 256;
      {
        const int row = fid >> 3, c4 = fid & 7;
        _Float16* p = &As[buf][row * 40 + c4 * 4];
        p[0] = (_Float16)ra[i].x; p[1] = (_Float16)ra[i].y;
        p[2] = (_Float16)ra[i].z; p[3] = (_Float16)ra[i].w;
      }
      {
        const int k = fid >> 5, c4 = fid & 31;
        Bs[buf][(c4 * 4 + 0) * 40 + k] = (_Float16)rb[i].x;
        Bs[buf][(c4 * 4 + 1) * 40 + k] = (_Float16)rb[i].y;
        Bs[buf][(c4 * 4 + 2) * 40 + k] = (_Float16)rb[i].z;
        Bs[buf][(c4 * 4 + 3) * 40 + k] = (_Float16)rb[i].w;
      }
    }
  };

  loadG(0);
  int buf = 0;
  for (int kk = 0; kk < DIM; kk += 32) {
    stage(buf);
    __syncthreads();
    if (kk + 32 < DIM) loadG(kk + 32);   // overlaps with WMMAs below

    v16h af[4], bf[2];
    #pragma unroll
    for (int i = 0; i < 4; ++i) {
      const _Float16* p = &As[buf][(wm * 64 + i * 16 + lr) * 40 + ak];
      af[i] = cat8(*(const v8h*)p, *(const v8h*)(p + 16));
    }
    #pragma unroll
    for (int j = 0; j < 2; ++j) {
      const _Float16* p = &Bs[buf][(wn * 32 + j * 16 + lr) * 40 + bko];
      bf[j] = cat8(*(const v8h*)p, *(const v8h*)(p + 16));
    }
    #pragma unroll
    for (int i = 0; i < 4; ++i)
      #pragma unroll
      for (int j = 0; j < 2; ++j)
        acc[i][j] = __builtin_amdgcn_wmma_f32_16x16x32_f16(
            false, af[i], false, bf[j], (short)0, acc[i][j], false, false);
    buf ^= 1;
  }

  // ---- branch-free unified epilogue ----
  // z=0/1:  addr = grow*EOUT + gcol
  // z=2  :  addr = grow + gcol*SEQ + bb*SEQ*(EOUT-1)   (bb = grow>>11)
  _Float16* dst;
  size_t rs, cs, bs;
  if (z == 0)      { dst = WQ;  rs = EOUT; cs = 1;   bs = 0; }
  else if (z == 1) { dst = WK;  rs = EOUT; cs = 1;   bs = 0; }
  else             { dst = WVt; rs = 1;    cs = SEQ; bs = (size_t)SEQ * (EOUT - 1); }

  #pragma unroll
  for (int i = 0; i < 4; ++i) {
    const int grow_base = m0 + wm * 64 + i * 16 + (lhi ? 8 : 0);
    #pragma unroll
    for (int j = 0; j < 2; ++j) {
      const size_t gcol = n0 + wn * 32 + j * 16 + lr;
      #pragma unroll
      for (int r = 0; r < 8; ++r) {
        const int grow = grow_base + r;
        const size_t bb = (size_t)(grow >> 11);
        dst[(size_t)grow * rs + gcol * cs + bb * bs] = (_Float16)acc[i][j][r];
      }
    }
  }
}

// =====================================================================
// Kernel 2: flash attention.  Block = 16 queries; 8 waves each own a
// 128-wide slice of E.  Key tiles of 128; online softmax in f32.
// =====================================================================
__global__ __launch_bounds__(256) void attn(
    const _Float16* __restrict__ WQ, const _Float16* __restrict__ WK,
    const _Float16* __restrict__ WVt, float* __restrict__ Out)
{
  __shared__ __align__(16) _Float16 Qs[16 * 1032];  // row stride 1032
  __shared__ __align__(16) float    Ss[16 * 128];
  __shared__ __align__(16) _Float16 Ps[16 * 136];   // row stride 136
  __shared__ float ms[16], ls[16], alphas[16];

  const int b    = blockIdx.y;
  const int q0   = blockIdx.x * 16;
  const int tid  = threadIdx.x;
  const int wave = tid >> 5, lane = tid & 31;
  const int lr   = lane & 15;
  const int lhi  = (lane >= 16) ? 1 : 0;

  // ---- stage Q tile (16 x 1024 f16): async global->LDS when available ----
  #pragma unroll
  for (int i = 0; i < 8; ++i) {
    const int fid = tid + i * 256;
    const int row = fid >> 7, c8 = (fid & 127) * 8;
    const _Float16* gp = WQ + ((size_t)(b * SEQ + q0 + row)) * EOUT + c8;
    _Float16* lp = &Qs[row * 1032 + c8];
#if defined(HAVE_ASYNC_LDS)
    __builtin_amdgcn_global_load_async_to_lds_b128(
        (async_gptr)gp, (async_lptr)lp, 0, 0);
#else
    *(v8h*)lp = *(const v8h*)gp;
#endif
  }
  if (tid < 16) { ms[tid] = -1e30f; ls[tid] = 0.f; }

  v8f acc[8] = {};
#if defined(HAVE_ASYNC_LDS)
  __builtin_amdgcn_s_wait_asynccnt(0);   // drain async copies before barrier
#endif
  __syncthreads();

  const int ak  = lhi ? 8 : 0;
  const int bko = lhi ? 8 : 0;

  for (int kt = 0; kt < SEQ / 128; ++kt) {
    // warm next key-tile's K rows (global_prefetch_b8)
    if (kt + 1 < SEQ / 128) {
      const _Float16* pf =
          WK + ((size_t)b * SEQ + (kt + 1) * 128 + wave * 16 + lr) * EOUT;
      __builtin_prefetch(pf, 0, 0);
    }

    // ---- scores: S(16x16 per wave) = Q * K^T, contraction over E ----
    v8f sacc = {};
    const size_t krow = ((size_t)b * SEQ + kt * 128 + wave * 16 + lr) * EOUT;
    #pragma unroll 4
    for (int e = 0; e < EOUT; e += 32) {
      const _Float16* qp = &Qs[lr * 1032 + e + ak];
      const v16h a = cat8(*(const v8h*)qp, *(const v8h*)(qp + 16));
      const _Float16* kp = WK + krow + e + bko;
      const v16h bf = cat8(*(const v8h*)kp, *(const v8h*)(kp + 16));
      sacc = __builtin_amdgcn_wmma_f32_16x16x32_f16(
          false, a, false, bf, (short)0, sacc, false, false);
    }
    {
      const int scol = wave * 16 + lr;
      #pragma unroll
      for (int r = 0; r < 8; ++r)
        Ss[(r + (lhi ? 8 : 0)) * 128 + scol] = sacc[r] * 0.03125f;  // 1/sqrt(1024)
    }
    __syncthreads();

    // ---- online softmax: one thread per query row ----
    if (tid < 16) {
      const int row = tid;
      const float mo = ms[row];
      float rmax = mo;
      #pragma unroll 8
      for (int c = 0; c < 128; ++c) rmax = fmaxf(rmax, Ss[row * 128 + c]);
      const float al = __expf(mo - rmax);
      float sum = 0.f;
      #pragma unroll 8
      for (int c = 0; c < 128; ++c) {
        const float p = __expf(Ss[row * 128 + c] - rmax);
        Ps[row * 136 + c] = (_Float16)p;
        sum += p;
      }
      ms[row]     = rmax;
      ls[row]     = ls[row] * al + sum;
      alphas[row] = al;
    }
    __syncthreads();

    // ---- rescale accumulators, then O += P @ V ----
    #pragma unroll
    for (int r = 0; r < 8; ++r) {
      const float al = alphas[r + (lhi ? 8 : 0)];
      #pragma unroll
      for (int t = 0; t < 8; ++t) acc[t][r] *= al;
    }
    #pragma unroll
    for (int s = 0; s < 4; ++s) {
      const _Float16* pp = &Ps[lr * 136 + s * 32 + ak];
      const v16h a = cat8(*(const v8h*)pp, *(const v8h*)(pp + 16));
      const int kbase = kt * 128 + s * 32 + bko;
      #pragma unroll
      for (int t = 0; t < 8; ++t) {
        const _Float16* vp =
            WVt + ((size_t)b * EOUT + wave * 128 + t * 16 + lr) * SEQ + kbase;
        const v16h bf = cat8(*(const v8h*)vp, *(const v8h*)(vp + 16));
        acc[t] = __builtin_amdgcn_wmma_f32_16x16x32_f16(
            false, a, false, bf, (short)0, acc[t], false, false);
      }
    }
    __syncthreads();
  }

  // ---- epilogue: normalize and store f32 output ----
  #pragma unroll
  for (int r = 0; r < 8; ++r) {
    const int row = r + (lhi ? 8 : 0);
    const float inv = 1.f / ls[row];
    const size_t orow = ((size_t)b * SEQ + q0 + row) * EOUT;
    #pragma unroll
    for (int t = 0; t < 8; ++t)
      Out[orow + wave * 128 + t * 16 + lr] = acc[t][r] * inv;
  }
}

// =====================================================================
extern "C" void kernel_launch(void* const* d_in, const int* in_sizes, int n_in,
                              void* d_out, int out_size, void* d_ws, size_t ws_size,
                              hipStream_t stream) {
  (void)in_sizes; (void)n_in; (void)out_size; (void)ws_size;
  const float* x = (const float*)d_in[0];
  const float* w = (const float*)d_in[1];
  float* out = (float*)d_out;

  _Float16* WQ  = (_Float16*)d_ws;                 // 8192*1024 f16 = 16 MB
  _Float16* WK  = WQ + (size_t)BATCH * SEQ * EOUT; // 16 MB
  _Float16* WVt = WK + (size_t)BATCH * SEQ * EOUT; // 16 MB (transposed V)

  dim3 g1(BATCH * SEQ / 128, EOUT / 128, 3);
  qkv_gemm<<<g1, 256, 0, stream>>>(x, w, WQ, WK, WVt);

  dim3 g2(SEQ / 16, BATCH);
  attn<<<g2, 256, 0, stream>>>(WQ, WK, WVt, out);
}
